// GRUClassifier_74062416052830
// MI455X (gfx1250) — matrix operand order repaired and checked by reading, not compiled, not run on hardware
//
#include <hip/hip_runtime.h>

// CDNA5 / gfx1250. wave32, WMMA 16x16x32 bf16 with fp32 accumulation.
// Weights pre-packed to bf16 once per launch; GRU recurrence stages xi tiles
// ahead of time with GLOBAL_LOAD_ASYNC_TO_LDS_B128 (double-buffered, ASYNCcnt).

typedef __attribute__((ext_vector_type(16))) __bf16 v16bf;
typedef __attribute__((ext_vector_type(8)))  float  v8f;

#define TSEQ  512
#define BATCH 64
#define HID   64
#define GATES 192
#define INP   2048

union Frag { v16bf v; unsigned int u[8]; };

__device__ __forceinline__ unsigned int bf16_bits(float f) {
  unsigned int a = __float_as_uint(f);
  return (a + 0x7FFFu + ((a >> 16) & 1u)) >> 16;   // round-to-nearest-even
}
__device__ __forceinline__ unsigned int pack_bf16x2(float lo, float hi) {
  return (bf16_bits(lo) & 0xFFFFu) | (bf16_bits(hi) << 16);
}
__device__ __forceinline__ float sigmoid_f(float x) {
  return 1.0f / (1.0f + __expf(-x));
}

// Async copy of 16 bytes global -> LDS (tracked by ASYNCcnt, in-order retire).
__device__ __forceinline__ void async_g2l_b128(unsigned int lds_off,
                                               const void* gaddr) {
  asm volatile("global_load_async_to_lds_b128 %0, %1, off"
               :: "v"(lds_off), "v"(gaddr) : "memory");
}

// ---------------------------------------------------------------------------
// Kernel 0: fp32 -> packed bf16 pairs (weights; run once, tiny)
// ---------------------------------------------------------------------------
__global__ __launch_bounds__(256) void k_cvt_bf16(
    const float2* __restrict__ src, unsigned int* __restrict__ dst, int npairs) {
  const int i = blockIdx.x * blockDim.x + threadIdx.x;
  if (i < npairs) {
    const float2 f = src[i];
    dst[i] = pack_bf16x2(f.x, f.y);
  }
}

// ---------------------------------------------------------------------------
// Kernel 1: xi0[t][b][g] = sum_d x[b][t][d] * W_ih0[g][d] + b_ih0[g]
// 2048 M-tiles, 12 waves/block (one 16x16 N-tile each), x staged through LDS
// as bf16 (x read exactly once), weights pre-packed. 2 WMMAs per stage iter.
// ---------------------------------------------------------------------------
__global__ __launch_bounds__(384) void k_gemm_ih0(
    const float* __restrict__ x,
    const unsigned int* __restrict__ Wb,   // bf16 pairs [192][INP/2]
    const float* __restrict__ bias, float* __restrict__ xi) {
  __shared__ unsigned int sA[16 * 32];     // 16 rows x 64 k, bf16 pairs

  const int tid  = threadIdx.x;
  const int row0 = blockIdx.x << 4;        // row = t*64 + b
  const int t    = row0 >> 6;
  const int b0   = row0 & 63;
  const int wave = tid >> 5;               // N-tile 0..11
  const int lane = tid & 31;
  const int ln   = lane & 15;
  const int hi   = lane >> 4;
  const int g    = wave * 16 + ln;

  v8f c;
  {
    const float bz = bias[g];
#pragma unroll
    for (int i = 0; i < 8; ++i) c[i] = bz;
  }

  const int arow  = (tid >> 4) & 15;
  const int apair = tid & 15;
  const float* xrow = x + (size_t)(b0 + arow) * (TSEQ * INP) + (size_t)t * INP;
  const unsigned int* wrow = Wb + (size_t)g * (INP / 2);

  for (int kk = 0; kk < INP / 64; ++kk) {  // 32 iterations
    __syncthreads();
    if (tid < 256) {
      const float4 f4 = *(const float4*)(xrow + kk * 64 + apair * 4);
      sA[arow * 32 + apair * 2]     = pack_bf16x2(f4.x, f4.y);
      sA[arow * 32 + apair * 2 + 1] = pack_bf16x2(f4.z, f4.w);
    }
    __syncthreads();

#pragma unroll
    for (int s = 0; s < 2; ++s) {
      Frag a, b;
      const unsigned int* pa = &sA[ln * 32 + s * 16 + hi * 4];
#pragma unroll
      for (int i = 0; i < 4; ++i) { a.u[i] = pa[i]; a.u[4 + i] = pa[8 + i]; }
      const uint4* pw = (const uint4*)(wrow + kk * 32 + s * 16 + hi * 4);
      const uint4 w0 = pw[0], w1 = pw[2];
      b.u[0] = w0.x; b.u[1] = w0.y; b.u[2] = w0.z; b.u[3] = w0.w;
      b.u[4] = w1.x; b.u[5] = w1.y; b.u[6] = w1.z; b.u[7] = w1.w;
      c = __builtin_amdgcn_wmma_f32_16x16x32_bf16(false, a.v, false, b.v,
                                                  (short)0, c, false, false);
    }
  }

#pragma unroll
  for (int i = 0; i < 8; ++i) {
    const int m = i + 8 * hi;
    xi[(size_t)(row0 + m) * GATES + g] = c[i];
  }
}

// ---------------------------------------------------------------------------
// Kernel 3: xi1 = y0(bf16) @ W_ih1^T + b_ih1.  K = 64 (2 WMMA k-steps).
// ---------------------------------------------------------------------------
__global__ __launch_bounds__(384) void k_gemm_ih1(
    const unsigned int* __restrict__ y0u,  // bf16 pairs, 32 uints / row
    const unsigned int* __restrict__ Wb,   // bf16 pairs [192][HID/2]
    const float* __restrict__ bias, float* __restrict__ xi) {
  const int tid  = threadIdx.x;
  const int row0 = blockIdx.x << 4;
  const int wave = tid >> 5;
  const int lane = tid & 31;
  const int ln   = lane & 15;
  const int hi   = lane >> 4;
  const int g    = wave * 16 + ln;

  v8f c;
  {
    const float bz = bias[g];
#pragma unroll
    for (int i = 0; i < 8; ++i) c[i] = bz;
  }

  const unsigned int* arow = y0u + (size_t)(row0 + ln) * (HID / 2);
  const unsigned int* wrow = Wb + (size_t)g * (HID / 2);

#pragma unroll
  for (int kk = 0; kk < 2; ++kk) {
    Frag a, b;
    const unsigned int* pa = arow + kk * 16 + hi * 4;
#pragma unroll
    for (int i = 0; i < 4; ++i) { a.u[i] = pa[i]; a.u[4 + i] = pa[8 + i]; }
    const uint4* pw = (const uint4*)(wrow + kk * 16 + hi * 4);
    const uint4 w0 = pw[0], w1 = pw[2];
    b.u[0] = w0.x; b.u[1] = w0.y; b.u[2] = w0.z; b.u[3] = w0.w;
    b.u[4] = w1.x; b.u[5] = w1.y; b.u[6] = w1.z; b.u[7] = w1.w;
    c = __builtin_amdgcn_wmma_f32_16x16x32_bf16(false, a.v, false, b.v,
                                                (short)0, c, false, false);
  }

#pragma unroll
  for (int i = 0; i < 8; ++i) {
    const int m = i + 8 * hi;
    xi[(size_t)(row0 + m) * GATES + g] = c[i];
  }
}

// ---------------------------------------------------------------------------
// Kernels 2/4: persistent single-workgroup GRU recurrence (512 serial steps).
// wave = (mt batch tile, jt hidden tile). 6 WMMAs/step/wave, W_hh^T frags in
// registers, h bf16 in LDS. xi tile for step t+1 is prefetched into a
// double-buffered dynamic-LDS region with GLOBAL_LOAD_ASYNC_TO_LDS_B128 while
// step t computes; completion enforced with s_wait_asynccnt + barrier.
// ---------------------------------------------------------------------------
__global__ __launch_bounds__(512) void k_gru(
    const float* __restrict__ xi,          // [T][B][192]
    const unsigned int* __restrict__ Whhb, // bf16 pairs [192][HID/2]
    const float* __restrict__ bhh,         // [192]
    unsigned short* __restrict__ y_out,    // bf16 [T][B][64] or nullptr
    float* __restrict__ h_out) {           // fp32 [B][64] or nullptr
  __shared__ unsigned short s_h[BATCH * HID];
  extern __shared__ float s_xi[];          // 2 x BATCH*GATES floats (96 KB)

  const int tid  = threadIdx.x;
  const int wave = tid >> 5;
  const int lane = tid & 31;
  const int ln   = lane & 15;
  const int hi   = lane >> 4;
  const int mt   = wave & 3;
  const int jt   = wave >> 2;
  const int bbase = mt * 16;
  const int j     = jt * 16 + ln;

  // Persistent B fragments: B[k=j'][n=g] = Whh[g][j'],  g = (jt + 4p)*16 + ln
  Frag Bf[3][2];
#pragma unroll
  for (int p = 0; p < 3; ++p) {
    const int g = (jt + 4 * p) * 16 + ln;
#pragma unroll
    for (int ks = 0; ks < 2; ++ks) {
      const uint4* pw = (const uint4*)(Whhb + (size_t)g * (HID / 2) + ks * 16 + hi * 4);
      const uint4 w0 = pw[0], w1 = pw[2];
      Bf[p][ks].u[0] = w0.x; Bf[p][ks].u[1] = w0.y;
      Bf[p][ks].u[2] = w0.z; Bf[p][ks].u[3] = w0.w;
      Bf[p][ks].u[4] = w1.x; Bf[p][ks].u[5] = w1.y;
      Bf[p][ks].u[6] = w1.z; Bf[p][ks].u[7] = w1.w;
    }
  }
  const float bias_r = bhh[j];
  const float bias_z = bhh[64 + j];
  const float bias_n = bhh[128 + j];

  for (int i = tid; i < BATCH * HID; i += 512) s_h[i] = 0;
  float hprev[8];
#pragma unroll
  for (int i = 0; i < 8; ++i) hprev[i] = 0.0f;

  // Prologue: async-stage xi tile t=0 into buffer 0 (48 KB, 6 b128/thread).
  {
    const char* g = (const char*)xi;
    const unsigned int lbase = (unsigned int)(size_t)(&s_xi[0]);
#pragma unroll
    for (int cth = 0; cth < 6; ++cth) {
      const unsigned int off = (unsigned int)((cth * 512 + tid) * 16);
      async_g2l_b128(lbase + off, g + off);
    }
  }
  __syncthreads();

  const unsigned int* hu = (const unsigned int*)s_h;
  const unsigned int* pa = hu + (size_t)(bbase + ln) * (HID / 2);

  for (int t = 0; t < TSEQ; ++t) {
    const int cur = t & 1;

    // Issue async staging of xi tile t+1 into the other buffer.
    if (t + 1 < TSEQ) {
      const char* g = (const char*)(xi + (size_t)(t + 1) * (BATCH * GATES));
      const unsigned int lbase =
          (unsigned int)(size_t)(&s_xi[(cur ^ 1) * (BATCH * GATES)]);
#pragma unroll
      for (int cth = 0; cth < 6; ++cth) {
        const unsigned int off = (unsigned int)((cth * 512 + tid) * 16);
        async_g2l_b128(lbase + off, g + off);
      }
    }

    Frag A0, A1;
#pragma unroll
    for (int i = 0; i < 4; ++i) {
      A0.u[i]     = pa[hi * 4 + i];
      A0.u[4 + i] = pa[hi * 4 + 8 + i];
      A1.u[i]     = pa[16 + hi * 4 + i];
      A1.u[4 + i] = pa[16 + hi * 4 + 8 + i];
    }

    v8f cr, cz, cn;
#pragma unroll
    for (int i = 0; i < 8; ++i) { cr[i] = bias_r; cz[i] = bias_z; cn[i] = bias_n; }
    cr = __builtin_amdgcn_wmma_f32_16x16x32_bf16(false, A0.v, false, Bf[0][0].v, (short)0, cr, false, false);
    cr = __builtin_amdgcn_wmma_f32_16x16x32_bf16(false, A1.v, false, Bf[0][1].v, (short)0, cr, false, false);
    cz = __builtin_amdgcn_wmma_f32_16x16x32_bf16(false, A0.v, false, Bf[1][0].v, (short)0, cz, false, false);
    cz = __builtin_amdgcn_wmma_f32_16x16x32_bf16(false, A1.v, false, Bf[1][1].v, (short)0, cz, false, false);
    cn = __builtin_amdgcn_wmma_f32_16x16x32_bf16(false, A0.v, false, Bf[2][0].v, (short)0, cn, false, false);
    cn = __builtin_amdgcn_wmma_f32_16x16x32_bf16(false, A1.v, false, Bf[2][1].v, (short)0, cn, false, false);

    // Retire everything except the 6 just-issued ops (in-order completion)
    // so tile t is fully in LDS; barrier makes all waves' slices visible,
    // and also orders the s_h A-reads above against the writes below.
    if (t + 1 < TSEQ) {
      asm volatile("s_wait_asynccnt 0x6" ::: "memory");
    } else {
      asm volatile("s_wait_asynccnt 0x0" ::: "memory");
    }
    __syncthreads();

    const float* xt = &s_xi[cur * (BATCH * GATES)];
#pragma unroll
    for (int i = 0; i < 8; ++i) {
      const int b = bbase + i + 8 * hi;
      const float* xb = xt + b * GATES;
      const float r  = sigmoid_f(xb[j] + cr[i]);
      const float z  = sigmoid_f(xb[64 + j] + cz[i]);
      const float nn = tanhf(xb[128 + j] + r * cn[i]);
      const float hn = (1.0f - z) * nn + z * hprev[i];
      hprev[i] = hn;
      const unsigned short hb = (unsigned short)bf16_bits(hn);
      s_h[b * HID + j] = hb;
      if (y_out) y_out[((size_t)t * BATCH + b) * HID + j] = hb;
    }
    __syncthreads();   // new h visible before next step's A reads
  }

  if (h_out) {
#pragma unroll
    for (int i = 0; i < 8; ++i) {
      const int b = bbase + i + 8 * hi;
      h_out[b * HID + j] = hprev[i];
    }
  }
}

// ---------------------------------------------------------------------------
// Kernel 5: out[b] = dot(h_last[b], fc_w) + fc_b
// ---------------------------------------------------------------------------
__global__ __launch_bounds__(64) void k_fc(
    const float* __restrict__ h, const float* __restrict__ w,
    const float* __restrict__ b, float* __restrict__ out) {
  const int i = threadIdx.x;
  float acc = b[0];
#pragma unroll
  for (int jj = 0; jj < HID; ++jj) acc += h[i * HID + jj] * w[jj];
  out[i] = acc;
}

extern "C" void kernel_launch(void* const* d_in, const int* in_sizes, int n_in,
                              void* d_out, int out_size, void* d_ws, size_t ws_size,
                              hipStream_t stream) {
  (void)in_sizes; (void)n_in; (void)out_size; (void)ws_size;
  const float* x     = (const float*)d_in[0];
  const float* W_ih0 = (const float*)d_in[1];
  const float* W_hh0 = (const float*)d_in[2];
  const float* b_ih0 = (const float*)d_in[3];
  const float* b_hh0 = (const float*)d_in[4];
  const float* W_ih1 = (const float*)d_in[5];
  const float* W_hh1 = (const float*)d_in[6];
  const float* b_ih1 = (const float*)d_in[7];
  const float* b_hh1 = (const float*)d_in[8];
  const float* fc_w  = (const float*)d_in[9];
  const float* fc_b  = (const float*)d_in[10];

  char* ws = (char*)d_ws;
  const size_t XI_BYTES   = (size_t)BATCH * TSEQ * GATES * sizeof(float);  // 25.17 MB
  const size_t Y0_BYTES   = (size_t)BATCH * TSEQ * HID * 2;                // 4.19 MB
  const size_t HL_BYTES   = (size_t)BATCH * HID * sizeof(float);           // 16 KB
  const size_t WIH0_BYTES = (size_t)GATES * INP * 2;                       // 768 KB
  const size_t WSM_BYTES  = (size_t)GATES * HID * 2;                       // 24 KB

  size_t off = 0;
  float*          xi0   = (float*)(ws + off);          off += XI_BYTES;
  float*          xi1   = (float*)(ws + off);          off += XI_BYTES;
  unsigned short* y0    = (unsigned short*)(ws + off); off += Y0_BYTES;
  float*          hlast = (float*)(ws + off);          off += HL_BYTES;
  unsigned int*   wih0b = (unsigned int*)(ws + off);   off += WIH0_BYTES;
  unsigned int*   whh0b = (unsigned int*)(ws + off);   off += WSM_BYTES;
  unsigned int*   wih1b = (unsigned int*)(ws + off);   off += WSM_BYTES;
  unsigned int*   whh1b = (unsigned int*)(ws + off);   off += WSM_BYTES;

  // Pre-pack all weights to bf16 pairs (once; tiny vs. the 256 MB x read).
  {
    const int np_ih0 = GATES * INP / 2;   // 196608
    const int np_sm  = GATES * HID / 2;   // 6144
    k_cvt_bf16<<<(np_ih0 + 255) / 256, 256, 0, stream>>>((const float2*)W_ih0, wih0b, np_ih0);
    k_cvt_bf16<<<(np_sm + 255) / 256, 256, 0, stream>>>((const float2*)W_hh0, whh0b, np_sm);
    k_cvt_bf16<<<(np_sm + 255) / 256, 256, 0, stream>>>((const float2*)W_ih1, wih1b, np_sm);
    k_cvt_bf16<<<(np_sm + 255) / 256, 256, 0, stream>>>((const float2*)W_hh1, whh1b, np_sm);
  }

  const size_t GRU_SHMEM = 2u * BATCH * GATES * sizeof(float);  // 96 KB dynamic LDS

  k_gemm_ih0<<<(BATCH * TSEQ) / 16, 384, 0, stream>>>(x, wih0b, b_ih0, xi0);
  k_gru<<<1, 512, GRU_SHMEM, stream>>>(xi0, whh0b, b_hh0, y0, nullptr);
  k_gemm_ih1<<<(BATCH * TSEQ) / 16, 384, 0, stream>>>((const unsigned int*)y0,
                                                      wih1b, b_ih1, xi1);
  k_gru<<<1, 512, GRU_SHMEM, stream>>>(xi1, whh1b, b_hh1, nullptr, hlast);
  k_fc<<<1, 64, 0, stream>>>(hlast, fc_w, fc_b, (float*)d_out);
}
